// SharedPrivateRouter_43843026157843
// MI455X (gfx1250) — matrix operand order repaired and checked by reading, not compiled, not run on hardware
//
#include <hip/hip_runtime.h>
#include <hip/hip_bf16.h>
#include <math.h>

#define D_MODEL  1024
#define D_HID    4096
#define N_TOK    4096      /* B*T = 4*1024 */
#define N_PRIV   8
#define N_SHRD   2
#define LDSTRIDE 40        /* halves per LDS tile row: 80B, 16B-aligned, conflict-free */

typedef __attribute__((ext_vector_type(16))) __bf16 v16bf;
typedef __attribute__((ext_vector_type(8)))  float  v8f;

union FragBF { uint4 q[2]; v16bf v; };

__device__ __forceinline__ unsigned short f2bf(float f) {
  unsigned int u = __float_as_uint(f);
  u += 0x7FFFu + ((u >> 16) & 1u);   // round-to-nearest-even
  return (unsigned short)(u >> 16);
}

// packed fp32 pair -> bf16 pair (lo in [15:0], hi in [31:16])
__device__ __forceinline__ unsigned int pack_bf(float lo, float hi) {
  __hip_bfloat162 h = __float22bfloat162_rn(make_float2(lo, hi));
  return *(unsigned int*)&h;
}

__device__ __forceinline__ float gelu_tanh(float v) {
  const float k0 = 0.7978845608028654f, k1 = 0.044715f;
  float t = tanhf(k0 * (v + k1 * v * v * v));
  return 0.5f * v * (1.0f + t);
}

// ---------------------------------------------------------------- init
__global__ void init_kernel(int* counts, float* probsum) {
  if (threadIdx.x < N_PRIV) { counts[threadIdx.x] = 0; probsum[threadIdx.x] = 0.f; }
}

// ---------------------------------------------------------------- router
__global__ __launch_bounds__(256)
void router_kernel(const float* __restrict__ x, const float* __restrict__ rw,
                   int* __restrict__ counts, float* __restrict__ probsum,
                   int* __restrict__ lists, float* __restrict__ wlists) {
  const int wv = threadIdx.x >> 5;
  const int lane = threadIdx.x & 31;
  const int t = blockIdx.x * 8 + wv;
  if (t >= N_TOK) return;

  float acc[N_PRIV];
#pragma unroll
  for (int e = 0; e < N_PRIV; ++e) acc[e] = 0.f;
  const float* xr = x + (size_t)t * D_MODEL;
  for (int d = lane; d < D_MODEL; d += 32) {
    float xv = xr[d];
    const float* r = rw + (size_t)d * N_PRIV;
#pragma unroll
    for (int e = 0; e < N_PRIV; ++e) acc[e] += xv * r[e];
  }
#pragma unroll
  for (int off = 16; off >= 1; off >>= 1) {
#pragma unroll
    for (int e = 0; e < N_PRIV; ++e) acc[e] += __shfl_xor(acc[e], off, 32);
  }

  if (lane == 0) {
    float m = acc[0];
#pragma unroll
    for (int e = 1; e < N_PRIV; ++e) m = fmaxf(m, acc[e]);
    float p[N_PRIV]; float s = 0.f;
#pragma unroll
    for (int e = 0; e < N_PRIV; ++e) { p[e] = __expf(acc[e] - m); s += p[e]; }
    float inv = 1.f / s;

    int i1 = 0;
#pragma unroll
    for (int e = 1; e < N_PRIV; ++e) if (p[e] > p[i1]) i1 = e;
    int i2 = (i1 == 0) ? 1 : 0;
#pragma unroll
    for (int e = 0; e < N_PRIV; ++e) if (e != i1 && p[e] > p[i2]) i2 = e;

    float p1 = p[i1] * inv, p2 = p[i2] * inv;
    float denom = fmaxf(p1 + p2, 1e-8f);
    float w1 = p1 / denom, w2 = p2 / denom;

#pragma unroll
    for (int e = 0; e < N_PRIV; ++e) atomicAdd(&probsum[e], p[e] * inv);

    int pos1 = atomicAdd(&counts[i1], 1);
    lists[i1 * N_TOK + pos1] = t;  wlists[i1 * N_TOK + pos1] = w1;
    int pos2 = atomicAdd(&counts[i2], 1);
    lists[i2 * N_TOK + pos2] = t;  wlists[i2 * N_TOK + pos2] = w2;
  }
}

// ---------------------------------------------------------------- x -> bf16
__global__ __launch_bounds__(256)
void cvt_kernel(const float* __restrict__ x, unsigned short* __restrict__ xbf) {
  size_t i = ((size_t)blockIdx.x * 256 + threadIdx.x) * 4;
  float4 v = *(const float4*)(x + i);
  uint2 r;
  r.x = pack_bf(v.x, v.y);
  r.y = pack_bf(v.z, v.w);
  *(uint2*)(xbf + i) = r;
}

// ---------------------------------------------------------------- tile helpers
// A tile: 128 rows x 32 halves; thread stages rows (ar, ar+64), 16B chunk ac.
// B tile: 32 k x 128 n; thread stages k-pairs (kk0, kk0+1) and (kk0+16, kk0+17)
//         for 4 consecutive n at nb, stored transposed [n][k] (pairs contiguous).
__device__ __forceinline__ void store_tiles(unsigned short* __restrict__ la,
                                            unsigned short* __restrict__ lb,
                                            int ar, int ac, int kk0, int nb,
                                            uint4 ra0, uint4 ra1,
                                            float4 w00, float4 w01,
                                            float4 w10, float4 w11) {
  *(uint4*)(la + ar * LDSTRIDE + ac * 8)        = ra0;
  *(uint4*)(la + (ar + 64) * LDSTRIDE + ac * 8) = ra1;
  *(unsigned int*)(lb + (nb + 0) * LDSTRIDE + kk0) = pack_bf(w00.x, w01.x);
  *(unsigned int*)(lb + (nb + 1) * LDSTRIDE + kk0) = pack_bf(w00.y, w01.y);
  *(unsigned int*)(lb + (nb + 2) * LDSTRIDE + kk0) = pack_bf(w00.z, w01.z);
  *(unsigned int*)(lb + (nb + 3) * LDSTRIDE + kk0) = pack_bf(w00.w, w01.w);
  *(unsigned int*)(lb + (nb + 0) * LDSTRIDE + kk0 + 16) = pack_bf(w10.x, w11.x);
  *(unsigned int*)(lb + (nb + 1) * LDSTRIDE + kk0 + 16) = pack_bf(w10.y, w11.y);
  *(unsigned int*)(lb + (nb + 2) * LDSTRIDE + kk0 + 16) = pack_bf(w10.z, w11.z);
  *(unsigned int*)(lb + (nb + 3) * LDSTRIDE + kk0 + 16) = pack_bf(w10.w, w11.w);
}

// One K-step: 2 A-frags x 4 B-frags -> 8 WMMAs (wave tile 32M x 64N).
__device__ __forceinline__ void mma_step(const unsigned short* __restrict__ la,
                                         const unsigned short* __restrict__ lb,
                                         int wm, int wn, int hi, int lr, v8f* acc) {
  FragBF A0, A1;
  const unsigned short* ap = la + (wm * 32 + lr) * LDSTRIDE + hi * 8;
  A0.q[0] = *(const uint4*)ap;
  A0.q[1] = *(const uint4*)(ap + 16);
  const unsigned short* ap1 = ap + 16 * LDSTRIDE;
  A1.q[0] = *(const uint4*)ap1;
  A1.q[1] = *(const uint4*)(ap1 + 16);
#pragma unroll
  for (int ns = 0; ns < 4; ++ns) {
    FragBF B;
    const unsigned short* bp = lb + (wn * 64 + ns * 16 + lr) * LDSTRIDE + hi * 8;
    B.q[0] = *(const uint4*)bp;
    B.q[1] = *(const uint4*)(bp + 16);
    acc[ns] = __builtin_amdgcn_wmma_f32_16x16x32_bf16(
        false, A0.v, false, B.v, (short)0, acc[ns], false, false);
    acc[4 + ns] = __builtin_amdgcn_wmma_f32_16x16x32_bf16(
        false, A1.v, false, B.v, (short)0, acc[4 + ns], false, false);
  }
}

// ---------------------------------------------------------------- GEMM1
// h = gelu(X @ W1 + b1) -> bf16.  Block tile 128M x 128N, double-buffered K=32.
__global__ __launch_bounds__(256)
void gemm1_kernel(const unsigned short* __restrict__ Xbf,
                  const float* __restrict__ W1, const float* __restrict__ B1,
                  unsigned short* __restrict__ Hout,
                  const int* __restrict__ rowlist, const int* __restrict__ countPtr,
                  int Mmax) {
  __shared__ unsigned short lds_a[2][128 * LDSTRIDE];
  __shared__ unsigned short lds_b[2][128 * LDSTRIDE];

  const int tid = threadIdx.x;
  const int count = countPtr ? *countPtr : Mmax;
  const int m0 = blockIdx.y * 128;
  if (m0 >= count) return;
  const int n0 = blockIdx.x * 128;

  const int ar = tid >> 2, ac = tid & 3;
  const int rg0 = m0 + ar, rg1 = m0 + ar + 64;
  const int tok0 = rowlist ? ((rg0 < count) ? rowlist[rg0] : 0) : rg0;
  const int tok1 = rowlist ? ((rg1 < count) ? rowlist[rg1] : 0) : rg1;
  const uint4* srcA0 = (const uint4*)(Xbf + (size_t)tok0 * D_MODEL) + ac;
  const uint4* srcA1 = (const uint4*)(Xbf + (size_t)tok1 * D_MODEL) + ac;

  const int kk0 = (tid >> 5) << 1;       // k rows kk0, kk0+1, kk0+16, kk0+17
  const int nb  = (tid & 31) << 2;       // 4 consecutive n
  const float* srcB = W1 + (size_t)kk0 * D_HID + (n0 + nb);

  const int wv = tid >> 5, lane = tid & 31;
  const int wm = wv & 3, wn = wv >> 2;
  const int hi = lane >> 4, lr = lane & 15;

  v8f zero = {};
  v8f acc[8];
#pragma unroll
  for (int i = 0; i < 8; ++i) acc[i] = zero;

  uint4 ra0 = *srcA0, ra1 = *srcA1;
  float4 w00 = *(const float4*)srcB;
  float4 w01 = *(const float4*)(srcB + D_HID);
  float4 w10 = *(const float4*)(srcB + 16 * D_HID);
  float4 w11 = *(const float4*)(srcB + 17 * D_HID);
  store_tiles(lds_a[0], lds_b[0], ar, ac, kk0, nb, ra0, ra1, w00, w01, w10, w11);
  __syncthreads();

  const int KSTEPS = D_MODEL / 32;
  for (int kt = 0; kt < KSTEPS; ++kt) {
    const bool more = (kt + 1) < KSTEPS;
    if (more) {                                     // prefetch next tile
      srcA0 += 4; srcA1 += 4;
      srcB += 32 * (size_t)D_HID;
      ra0 = *srcA0; ra1 = *srcA1;
      w00 = *(const float4*)srcB;
      w01 = *(const float4*)(srcB + D_HID);
      w10 = *(const float4*)(srcB + 16 * D_HID);
      w11 = *(const float4*)(srcB + 17 * D_HID);
    }
    mma_step(lds_a[kt & 1], lds_b[kt & 1], wm, wn, hi, lr, acc);
    if (more)
      store_tiles(lds_a[(kt + 1) & 1], lds_b[(kt + 1) & 1],
                  ar, ac, kk0, nb, ra0, ra1, w00, w01, w10, w11);
    __syncthreads();
  }

#pragma unroll
  for (int rs = 0; rs < 2; ++rs) {
#pragma unroll
    for (int ns = 0; ns < 4; ++ns) {
      int n = n0 + wn * 64 + ns * 16 + lr;
      float bias = B1[n];
#pragma unroll
      for (int r = 0; r < 8; ++r) {
        int mloc = m0 + wm * 32 + rs * 16 + r + hi * 8;
        if (mloc < count) {
          float val = gelu_tanh(acc[rs * 4 + ns][r] + bias);
          Hout[(size_t)mloc * D_HID + n] = f2bf(val);
        }
      }
    }
  }
}

// ---------------------------------------------------------------- GEMM2
// out[token] (+)= w * (H @ W2 + b2).  K = D_HID, double-buffered.
__global__ __launch_bounds__(256)
void gemm2_kernel(const unsigned short* __restrict__ Hbf,
                  const float* __restrict__ W2, const float* __restrict__ B2,
                  float* __restrict__ Out,
                  const int* __restrict__ rowlist, const float* __restrict__ rowweight,
                  const int* __restrict__ countPtr,
                  int Mmax, float scale, int accumulate) {
  __shared__ unsigned short lds_a[2][128 * LDSTRIDE];
  __shared__ unsigned short lds_b[2][128 * LDSTRIDE];
  __shared__ int   toks[128];
  __shared__ float wgt[128];

  const int tid = threadIdx.x;
  const int count = countPtr ? *countPtr : Mmax;
  const int m0 = blockIdx.y * 128;
  if (m0 >= count) return;
  const int n0 = blockIdx.x * 128;

  if (tid < 128) {                                 // epilogue scatter metadata
    int r = m0 + tid;
    toks[tid] = rowlist ? ((r < count) ? rowlist[r] : 0) : r;
    wgt[tid]  = rowweight ? ((r < count) ? rowweight[r] : 0.f) : scale;
  }

  const int ar = tid >> 2, ac = tid & 3;
  const uint4* srcA0 = (const uint4*)(Hbf + (size_t)(m0 + ar) * D_HID) + ac;
  const uint4* srcA1 = (const uint4*)(Hbf + (size_t)(m0 + ar + 64) * D_HID) + ac;
  const int kk0 = (tid >> 5) << 1;
  const int nb  = (tid & 31) << 2;
  const float* srcB = W2 + (size_t)kk0 * D_MODEL + (n0 + nb);

  const int wv = tid >> 5, lane = tid & 31;
  const int wm = wv & 3, wn = wv >> 2;
  const int hi = lane >> 4, lr = lane & 15;

  v8f zero = {};
  v8f acc[8];
#pragma unroll
  for (int i = 0; i < 8; ++i) acc[i] = zero;

  uint4 ra0 = *srcA0, ra1 = *srcA1;
  float4 w00 = *(const float4*)srcB;
  float4 w01 = *(const float4*)(srcB + D_MODEL);
  float4 w10 = *(const float4*)(srcB + 16 * D_MODEL);
  float4 w11 = *(const float4*)(srcB + 17 * D_MODEL);
  store_tiles(lds_a[0], lds_b[0], ar, ac, kk0, nb, ra0, ra1, w00, w01, w10, w11);
  __syncthreads();

  const int KSTEPS = D_HID / 32;
  for (int kt = 0; kt < KSTEPS; ++kt) {
    const bool more = (kt + 1) < KSTEPS;
    if (more) {
      srcA0 += 4; srcA1 += 4;
      srcB += 32 * (size_t)D_MODEL;
      ra0 = *srcA0; ra1 = *srcA1;
      w00 = *(const float4*)srcB;
      w01 = *(const float4*)(srcB + D_MODEL);
      w10 = *(const float4*)(srcB + 16 * D_MODEL);
      w11 = *(const float4*)(srcB + 17 * D_MODEL);
    }
    mma_step(lds_a[kt & 1], lds_b[kt & 1], wm, wn, hi, lr, acc);
    if (more)
      store_tiles(lds_a[(kt + 1) & 1], lds_b[(kt + 1) & 1],
                  ar, ac, kk0, nb, ra0, ra1, w00, w01, w10, w11);
    __syncthreads();
  }

#pragma unroll
  for (int rs = 0; rs < 2; ++rs) {
#pragma unroll
    for (int ns = 0; ns < 4; ++ns) {
      int n = n0 + wn * 64 + ns * 16 + lr;
      float bias = B2[n];
#pragma unroll
      for (int r = 0; r < 8; ++r) {
        int li = wm * 32 + rs * 16 + r + hi * 8;
        int mloc = m0 + li;
        if (mloc < count) {
          float val = wgt[li] * (acc[rs * 4 + ns][r] + bias);
          size_t o = (size_t)toks[li] * D_MODEL + n;
          if (accumulate) Out[o] += val; else Out[o] = val;
        }
      }
    }
  }
}

// ---------------------------------------------------------------- aux loss
__global__ void aux_kernel(const int* counts, const float* probsum, float* out) {
  if (threadIdx.x == 0) {
    float s = 0.f;
    for (int e = 0; e < N_PRIV; ++e) {
      float f = (float)counts[e] / (float)N_TOK;
      float p = probsum[e] / (float)N_TOK;
      s += f * p;
    }
    out[(size_t)N_TOK * D_MODEL] = 0.01f * (float)N_PRIV * s;
  }
}

// ---------------------------------------------------------------- launch
extern "C" void kernel_launch(void* const* d_in, const int* in_sizes, int n_in,
                              void* d_out, int out_size, void* d_ws, size_t ws_size,
                              hipStream_t stream) {
  const float* x   = (const float*)d_in[0];
  const float* sw1 = (const float*)d_in[1];
  const float* sb1 = (const float*)d_in[2];
  const float* sw2 = (const float*)d_in[3];
  const float* sb2 = (const float*)d_in[4];
  const float* pw1 = (const float*)d_in[5];
  const float* pb1 = (const float*)d_in[6];
  const float* pw2 = (const float*)d_in[7];
  const float* pb2 = (const float*)d_in[8];
  const float* rw  = (const float*)d_in[9];
  float* out = (float*)d_out;

  char* ws = (char*)d_ws;
  int*   counts  = (int*)ws;
  float* probsum = (float*)(ws + 64);
  int*   lists   = (int*)(ws + 128);
  float* wlists  = (float*)(ws + 128 + sizeof(int) * N_PRIV * N_TOK);
  unsigned short* xbf  = (unsigned short*)(ws + (size_t)(1 << 20));              // +1MB
  unsigned short* hbuf = (unsigned short*)(ws + (size_t)(1 << 20) +
                                           (size_t)N_TOK * D_MODEL * 2);         // +9MB

  dim3 blk(256);
  init_kernel<<<dim3(1), dim3(32), 0, stream>>>(counts, probsum);
  router_kernel<<<dim3(N_TOK / 8), blk, 0, stream>>>(x, rw, counts, probsum,
                                                     lists, wlists);
  cvt_kernel<<<dim3((N_TOK * D_MODEL) / 1024), blk, 0, stream>>>(x, xbf);

  dim3 g1(D_HID / 128, N_TOK / 128);    // 32 x 32 blocks
  dim3 g2(D_MODEL / 128, N_TOK / 128);  //  8 x 32 blocks

  for (int s = 0; s < N_SHRD; ++s) {
    gemm1_kernel<<<g1, blk, 0, stream>>>(
        xbf, sw1 + (size_t)s * D_MODEL * D_HID, sb1 + (size_t)s * D_HID,
        hbuf, nullptr, nullptr, N_TOK);
    gemm2_kernel<<<g2, blk, 0, stream>>>(
        hbuf, sw2 + (size_t)s * D_HID * D_MODEL, sb2 + (size_t)s * D_MODEL,
        out, nullptr, nullptr, nullptr, N_TOK, 1.0f / N_SHRD, s == 0 ? 0 : 1);
  }
  for (int e = 0; e < N_PRIV; ++e) {
    gemm1_kernel<<<g1, blk, 0, stream>>>(
        xbf, pw1 + (size_t)e * D_MODEL * D_HID, pb1 + (size_t)e * D_HID,
        hbuf, lists + (size_t)e * N_TOK, counts + e, N_TOK);
    gemm2_kernel<<<g2, blk, 0, stream>>>(
        hbuf, pw2 + (size_t)e * D_HID * D_MODEL, pb2 + (size_t)e * D_MODEL,
        out, lists + (size_t)e * N_TOK, wlists + (size_t)e * N_TOK, counts + e,
        N_TOK, 1.0f, 1);
  }
  aux_kernel<<<dim3(1), dim3(32), 0, stream>>>(counts, probsum, out);
}
